// RecursiveLSTM_55628416417933
// MI455X (gfx1250) — matrix operand order repaired and checked by reading, not compiled, not run on hardware
//
#include <hip/hip_runtime.h>
#include <hip/hip_bf16.h>
#include <cmath>

typedef _Float16 v16h __attribute__((ext_vector_type(16)));
typedef float    v8f  __attribute__((ext_vector_type(8)));

#define HH   50      // hidden size
#define TT   96      // window length
#define G4   200     // 4*H gates
#define NT   13      // ceil(200/16) N-tiles
#define KPAD 64      // K padded to 2x32 for wmma 16x16x32
#define BW   16      // batch rows per wave (WMMA M)

#define LOG2E 1.4426950408889634f

// Fast transcendentals: keep the activation chain on hardware TRANS32 ops
// (v_exp_f32 / v_rcp_f32 / v_tanh_f32) instead of precise-division sequences.
__device__ __forceinline__ float fast_rcp(float x) {
    return __builtin_amdgcn_rcpf(x);
}
__device__ __forceinline__ float fast_exp(float x) {
    return __builtin_amdgcn_exp2f(x * LOG2E);
}
__device__ __forceinline__ float fast_sigmoid(float x) {
    return fast_rcp(1.0f + fast_exp(-x));
}
__device__ __forceinline__ float fast_tanh(float x) {
#if __has_builtin(__builtin_amdgcn_tanhf)
    return __builtin_amdgcn_tanhf(x);          // v_tanh_f32 (CDNA5 TRANS op)
#elif __has_builtin(__builtin_amdgcn_tanh_f32)
    return __builtin_amdgcn_tanh_f32(x);
#else
    float e = fast_exp(-2.0f * x);             // tanh = (1-e^-2x)/(1+e^-2x)
    return (1.0f - e) * fast_rcp(1.0f + e);
#endif
}

__global__ __launch_bounds__(32)
void lstm_forecast_kernel(const float* __restrict__ x,
                          const float* __restrict__ W_ih,
                          const float* __restrict__ W_hh,
                          const float* __restrict__ b_ih,
                          const float* __restrict__ b_hh,
                          const float* __restrict__ W_fc,
                          const float* __restrict__ b_fc,
                          float* __restrict__ out,
                          int B, int NP)
{
    // ---- LDS (single wave per block, ~52.6 KB) ----
    __shared__ __align__(32) _Float16 Bfrag[NT][2][32][16]; // W_hh^T in WMMA B-fragment layout
    __shared__ __align__(16) _Float16 h16[BW][KPAD];        // h state, f16, A-fragment friendly
    __shared__ float c_s[BW][HH];                           // c state
    __shared__ float gat[BW][NT * 16];                      // activated gates staging
    __shared__ float x_s[BW][TT];                           // input window
    __shared__ float pr_s[BW][16];                          // predictions so far
    __shared__ float xt_s[BW];                              // current-step scalar input
    __shared__ float wfc_s[HH];                             // FC weights

    const int lane = threadIdx.x;     // 0..31
    const int n    = lane & 15;       // WMMA column / row-in-half
    const int half = lane >> 4;       // which 16-lane half
    const int base = blockIdx.x * BW; // first batch row owned by this wave

    // ================= one-time setup =================
    for (int idx = lane; idx < BW * TT; idx += 32) {
        int r = idx / TT, t = idx % TT;
        x_s[r][t] = x[(size_t)(base + r) * TT + t];
    }
    // B-fragment layout: element i of lane half h holds K = kit*32 + 8h + i (i<8),
    // K = kit*32 + 16 + 8h + (i-8) (i>=8) of column n (matches A order).
    for (int t13 = 0; t13 < NT; ++t13)
        for (int kit = 0; kit < 2; ++kit)
            #pragma unroll
            for (int i = 0; i < 16; ++i) {
                int j = t13 * 16 + n;             // gate index (N)
                int k = kit * 32 + (i < 8 ? half * 8 + i : 16 + half * 8 + (i - 8)); // hidden index (K)
                float v = (j < G4 && k < HH) ? W_hh[j * HH + k] : 0.0f;
                Bfrag[t13][kit][lane][i] = (_Float16)v;
            }
    // per-lane per-tile constants: rank-1 input weight and fused bias
    float wih_r[NT], bs_r[NT];
    #pragma unroll
    for (int t13 = 0; t13 < NT; ++t13) {
        int j = t13 * 16 + n;
        wih_r[t13] = (j < G4) ? W_ih[j] : 0.0f;
        bs_r[t13]  = (j < G4) ? (b_ih[j] + b_hh[j]) : 0.0f;
    }
    for (int k = lane; k < HH; k += 32) wfc_s[k] = W_fc[k];
    const float bfc = b_fc[0];
    __syncthreads();

    const int r   = lane & 15;  // recombination row
    const int pt  = lane >> 4;  // recombination half (0/1)
    const int kk0 = pt * 25;    // each half handles 25 hidden channels

    for (int p = 0; p < NP; ++p) {
        // zero (h, c) for this prediction pass (K-padding of h16 stays zero)
        for (int idx = lane; idx < BW * KPAD; idx += 32)
            ((_Float16*)h16)[idx] = (_Float16)0.0f;
        for (int idx = lane; idx < BW * HH; idx += 32)
            ((float*)c_s)[idx] = 0.0f;
        __syncthreads();

        for (int t = 0; t < TT; ++t) {
            // sliding-window input: real x for t < TT-p, else earlier predictions
            if (lane < BW)
                xt_s[lane] = (t < TT - p) ? x_s[lane][p + t]
                                          : pr_s[lane][t - (TT - p)];
            __syncthreads();

            // A fragments from h16: a[i] = h16[m][ (i<8?i:i+8) + half*8 (+32 for a1) ]
            v16h a0, a1;
            #pragma unroll
            for (int i = 0; i < 8; ++i) {
                a0[i]     = h16[n][half * 8 + i];
                a0[8 + i] = h16[n][16 + half * 8 + i];
                a1[i]     = h16[n][32 + half * 8 + i];
                a1[8 + i] = h16[n][48 + half * 8 + i];
            }
            float xs[8];
            #pragma unroll
            for (int i = 0; i < 8; ++i) xs[i] = xt_s[i + half * 8];

            // gates = h @ W_hh^T  via 13 N-tiles x 2 K-chunks of WMMA f16->f32
            for (int t13 = 0; t13 < NT; ++t13) {
                v16h b0 = *reinterpret_cast<const v16h*>(&Bfrag[t13][0][lane][0]);
                v16h b1 = *reinterpret_cast<const v16h*>(&Bfrag[t13][1][lane][0]);
                v8f acc = {};
                acc = __builtin_amdgcn_wmma_f32_16x16x32_f16(
                          false, a0, false, b0, (short)0, acc, false, false);
                acc = __builtin_amdgcn_wmma_f32_16x16x32_f16(
                          false, a1, false, b1, (short)0, acc, false, false);

                int j = t13 * 16 + n; // gate column
                if (j < G4) {
                    const bool is_tanh = (j >= 2 * HH) && (j < 3 * HH); // g-gate
                    const float wih = wih_r[t13], bs = bs_r[t13];
                    #pragma unroll
                    for (int i = 0; i < 8; ++i) { // D row = i + half*8
                        float pre = acc[i] + xs[i] * wih + bs;
                        float av  = is_tanh ? fast_tanh(pre) : fast_sigmoid(pre);
                        gat[i + half * 8][j] = av;
                    }
                }
            }
            __syncthreads();

            // cell update: c' = f*c + i*g ; h' = o*tanh(c')  (2 lanes per row)
            for (int kk = kk0; kk < kk0 + 25; ++kk) {
                float iv = gat[r][kk];
                float fv = gat[r][HH + kk];
                float gv = gat[r][2 * HH + kk];
                float ov = gat[r][3 * HH + kk];
                float cv = fv * c_s[r][kk] + iv * gv;
                c_s[r][kk] = cv;
                h16[r][kk] = (_Float16)(ov * fast_tanh(cv));
            }
            __syncthreads();
        }

        // FC head: y = h_T . W_fc + b_fc
        float part = 0.0f;
        for (int kk = kk0; kk < kk0 + 25; ++kk)
            part += (float)h16[r][kk] * wfc_s[kk];
        float tot = part + __shfl_xor(part, 16, 32);
        if (pt == 0) {
            float y = tot + bfc;
            pr_s[r][p] = y;
            out[(size_t)(base + r) * NP + p] = y;
        }
        __syncthreads();
    }
}

extern "C" void kernel_launch(void* const* d_in, const int* in_sizes, int n_in,
                              void* d_out, int out_size, void* d_ws, size_t ws_size,
                              hipStream_t stream) {
    (void)n_in; (void)d_ws; (void)ws_size;
    const float* x    = (const float*)d_in[0];
    const float* W_ih = (const float*)d_in[1];
    const float* W_hh = (const float*)d_in[2];
    const float* b_ih = (const float*)d_in[3];
    const float* b_hh = (const float*)d_in[4];
    const float* W_fc = (const float*)d_in[5];
    const float* b_fc = (const float*)d_in[6];

    const int B  = in_sizes[0] / TT;   // x is [B, 96, 1]
    const int NP = out_size / B;       // output is [B, num_pred]

    dim3 grid(B / BW), block(32);
    hipLaunchKernelGGL(lstm_forecast_kernel, grid, block, 0, stream,
                       x, W_ih, W_hh, b_ih, b_hh, W_fc, b_fc,
                       (float*)d_out, B, NP);
}
